// NODEBlock_30202210025634
// MI455X (gfx1250) — matrix operand order repaired and checked by reading, not compiled, not run on hardware
//
#include <hip/hip_runtime.h>
#include <hip/hip_bf16.h>

// ---------------- problem constants ----------------
#define BATCH      32768
#define INPUT_DIM  256
#define NUM_TREES  64
#define TREE_DEPTH 6
#define NUM_LEAVES 64
#define NCOLS      (NUM_TREES * TREE_DEPTH)   // 384 decision columns
#define DSTRIDE    (NCOLS + 1)                // padded LDS row stride (bank-conflict free)
#define BN_EPS     1e-5f

typedef __attribute__((ext_vector_type(16))) __bf16 v16bf;
typedef __attribute__((ext_vector_type(8)))  float  v8f;

// ---------------- WMMA fragment loaders (CDNA5 wave32 layouts) ----------------
// A 16x32 bf16 (ISA 7.12.2): lanes 0-15: VGPR0-3 = K0..7,  VGPR4-7 = K16..23
//                            lanes16-31: VGPR0-3 = K8..15, VGPR4-7 = K24..31
// => per lane: two 8-element chunks at (half*8) and (half*8 + 16)
__device__ __forceinline__ v16bf frag_a(const __bf16* p, int half) {
  union { uint4 q[2]; v16bf v; } u;
  const __bf16* c = p + half * 8;
  u.q[0] = *(const uint4*)(c);
  u.q[1] = *(const uint4*)(c + 16);
  return u.v;
}
// B 32x16 bf16: lanes 0-15: K=0..7,16..23 (col N=lane), lanes16-31: K=8..15,24..31
// => per lane: two 8-element chunks at (half*8) and (half*8 + 16)
__device__ __forceinline__ v16bf frag_b(const __bf16* p, int half) {
  union { uint4 q[2]; v16bf v; } u;
  const __bf16* c = p + half * 8;
  u.q[0] = *(const uint4*)(c);
  u.q[1] = *(const uint4*)(c + 16);
  return u.v;
}

// ---------------- Kernel 0: one-shot f32 -> bf16 weight conversion ----------------
__global__ __launch_bounds__(256) void cvt_bf16_kernel(const float* __restrict__ in,
                                                       __bf16* __restrict__ outp, int n) {
  const int i = blockIdx.x * 256 + threadIdx.x;
  if (i < n) outp[i] = (__bf16)in[i];
}

// ---------------- Kernel 1 (fused): GEMM + sigmoid + tree fold + BN ----------------
// Phase A: 64x384 decision GEMM (K=256) with x-tile staged in LDS as bf16;
//          wave w owns 3 n-tiles x 4 m-tiles -> 12 WMMAs per K-step.
// Phase B: sigmoid(dec) written to LDS (64 x 385 f32, padded stride).
// Phase C: 6-level soft-tree fold + BatchNorm, bf16 tnorm out (only 128 B/row
//          leaves the WGP instead of a 50 MB dec round-trip through L2/HBM).
__global__ __launch_bounds__(256) void dec_tree_kernel(
    const float*  __restrict__ x,      // [BATCH, 256]
    const __bf16* __restrict__ Wd,     // [384, 256] bf16 (pre-converted)
    const float*  __restrict__ bdec,   // [384]
    const float*  __restrict__ leaf,   // [64, 64]
    const float*  __restrict__ gamma, const float* __restrict__ beta,
    const float*  __restrict__ mean,  const float* __restrict__ var,
    __bf16* __restrict__ tnorm)        // [BATCH, 64] bf16
{
  extern __shared__ char smem_raw[];
  float*  DecS = (float*)smem_raw;     // 64 * 385 f32 = 96.25 KB (phase B/C)
  __bf16* As   = (__bf16*)smem_raw;    // 64 * 256 bf16 = 32 KB (phase A, aliased)

  const int tid   = threadIdx.x;
  const int mbase = blockIdx.x * 64;

  // ---- Phase A0: cooperative stage 64x256 f32 -> bf16 (coalesced float4) ----
  const float* xt = x + (size_t)mbase * INPUT_DIM;
#pragma unroll
  for (int it = 0; it < 16; ++it) {
    const int e = it * 1024 + tid * 4;
    const float4 f = *(const float4*)(xt + e);
    union { __bf16 h[4]; uint2 u; } c;
    c.h[0] = (__bf16)f.x; c.h[1] = (__bf16)f.y;
    c.h[2] = (__bf16)f.z; c.h[3] = (__bf16)f.w;
    *(uint2*)(As + e) = c.u;
  }
  __syncthreads();

  const int lane = tid & 31, wave = tid >> 5;
  const int half = lane >> 4, l16 = lane & 15;

  int bcol[3];
  const __bf16* bptr[3];
#pragma unroll
  for (int i = 0; i < 3; ++i) {
    bcol[i] = (wave * 3 + i) * 16 + l16;
    bptr[i] = Wd + (size_t)bcol[i] * INPUT_DIM;
  }

  // ---- Phase A1: K-loop, 12 WMMAs per step ----
  v8f acc[3][4];
#pragma unroll
  for (int i = 0; i < 3; ++i)
#pragma unroll
    for (int mt = 0; mt < 4; ++mt) acc[i][mt] = (v8f){};

#pragma unroll
  for (int ks = 0; ks < INPUT_DIM / 32; ++ks) {
    v16bf a[4];
#pragma unroll
    for (int mt = 0; mt < 4; ++mt)
      a[mt] = frag_a(As + (mt * 16 + l16) * INPUT_DIM + ks * 32, half);
#pragma unroll
    for (int i = 0; i < 3; ++i) {
      const v16bf b = frag_b(bptr[i] + ks * 32, half);
#pragma unroll
      for (int mt = 0; mt < 4; ++mt)
        acc[i][mt] = __builtin_amdgcn_wmma_f32_16x16x32_bf16(
            false, a[mt], false, b, (short)0, acc[i][mt], false, false);
    }
  }
  __syncthreads();   // all waves done reading As before DecS overwrites it

  // ---- Phase B: sigmoid(dec + bias) -> LDS (C layout: VGPR j -> M=j+8*half, N=l16) ----
#pragma unroll
  for (int i = 0; i < 3; ++i) {
    const float bias = bdec[bcol[i]];
#pragma unroll
    for (int mt = 0; mt < 4; ++mt)
#pragma unroll
      for (int j = 0; j < 8; ++j) {
        const int r = mt * 16 + j + 8 * half;
        float v = acc[i][mt][j] + bias;
        v = 1.0f / (1.0f + __expf(-v));           // sigmoid
        DecS[r * DSTRIDE + bcol[i]] = v;
      }
  }
  __syncthreads();

  // ---- Phase C: soft-tree fold + BN.  thread = tree t, 16 rows each ----
  const int t  = tid & 63;
  const int rq = tid >> 6;

  float lv[NUM_LEAVES];
  const float* lp = leaf + t * NUM_LEAVES;
#pragma unroll
  for (int i = 0; i < NUM_LEAVES; ++i) lv[i] = lp[i];

  const float sc = gamma[t] * rsqrtf(var[t] + BN_EPS);
  const float sh = beta[t] - mean[t] * sc;

  for (int r = rq; r < 64; r += 4) {
    const float* dp = DecS + r * DSTRIDE + t * TREE_DEPTH;
    float d[TREE_DEPTH];
#pragma unroll
    for (int k = 0; k < TREE_DEPTH; ++k) d[k] = dp[k];

    float w[32];
#pragma unroll
    for (int j = 0; j < 32; ++j) w[j] = lv[2 * j] + d[5] * (lv[2 * j + 1] - lv[2 * j]);
#pragma unroll
    for (int j = 0; j < 16; ++j) w[j] = w[2 * j] + d[4] * (w[2 * j + 1] - w[2 * j]);
#pragma unroll
    for (int j = 0; j < 8; ++j)  w[j] = w[2 * j] + d[3] * (w[2 * j + 1] - w[2 * j]);
#pragma unroll
    for (int j = 0; j < 4; ++j)  w[j] = w[2 * j] + d[2] * (w[2 * j + 1] - w[2 * j]);
#pragma unroll
    for (int j = 0; j < 2; ++j)  w[j] = w[2 * j] + d[1] * (w[2 * j + 1] - w[2 * j]);
    const float res = w[0] + d[0] * (w[1] - w[0]);

    tnorm[(size_t)(mbase + r) * NUM_TREES + t] = (__bf16)(res * sc + sh);
  }
}

// ---------------- Kernel 2: out = tnorm @ fc_w^T + fc_b ----------------
// M=BATCH, N=256, K=64.  grid=BATCH/64, block=256 (8 waves).
// 64x64 bf16 A-tile staged in LDS; wave w owns 2 n-tiles x 4 m-tiles.
__global__ __launch_bounds__(256) void out_gemm_kernel(
    const __bf16* __restrict__ tnorm,  // [BATCH, 64] bf16
    const __bf16* __restrict__ fcw,    // [256, 64] bf16 (pre-converted)
    const float*  __restrict__ fcb,    // [256]
    float* __restrict__ out)           // [BATCH, 256]
{
  __shared__ alignas(16) __bf16 As[64 * NUM_TREES];   // 8 KB
  const int tid   = threadIdx.x;
  const int mbase = blockIdx.x * 64;

  const uint4* src = (const uint4*)(tnorm + (size_t)mbase * NUM_TREES);
  uint4* dst = (uint4*)As;
#pragma unroll
  for (int it = 0; it < 2; ++it) dst[it * 256 + tid] = src[it * 256 + tid];
  __syncthreads();

  const int lane = tid & 31, wave = tid >> 5;
  const int half = lane >> 4, l16 = lane & 15;

  int col[2];
  const __bf16* bptr[2];
#pragma unroll
  for (int i = 0; i < 2; ++i) {
    col[i]  = (wave * 2 + i) * 16 + l16;
    bptr[i] = fcw + (size_t)col[i] * NUM_TREES;
  }

  v8f acc[2][4];
#pragma unroll
  for (int i = 0; i < 2; ++i)
#pragma unroll
    for (int mt = 0; mt < 4; ++mt) acc[i][mt] = (v8f){};

#pragma unroll
  for (int ks = 0; ks < NUM_TREES / 32; ++ks) {
    v16bf a[4];
#pragma unroll
    for (int mt = 0; mt < 4; ++mt)
      a[mt] = frag_a(As + (mt * 16 + l16) * NUM_TREES + ks * 32, half);
#pragma unroll
    for (int i = 0; i < 2; ++i) {
      const v16bf b = frag_b(bptr[i] + ks * 32, half);
#pragma unroll
      for (int mt = 0; mt < 4; ++mt)
        acc[i][mt] = __builtin_amdgcn_wmma_f32_16x16x32_bf16(
            false, a[mt], false, b, (short)0, acc[i][mt], false, false);
    }
  }

#pragma unroll
  for (int i = 0; i < 2; ++i) {
    const float bias = fcb[col[i]];
#pragma unroll
    for (int mt = 0; mt < 4; ++mt)
#pragma unroll
      for (int j = 0; j < 8; ++j) {
        const int m = mbase + mt * 16 + j + 8 * half;
        out[(size_t)m * INPUT_DIM + col[i]] = acc[i][mt][j] + bias;
      }
  }
}

// ---------------- launcher ----------------
extern "C" void kernel_launch(void* const* d_in, const int* in_sizes, int n_in,
                              void* d_out, int out_size, void* d_ws, size_t ws_size,
                              hipStream_t stream) {
  const float* x     = (const float*)d_in[0];
  const float* W_dec = (const float*)d_in[1];
  const float* b_dec = (const float*)d_in[2];
  const float* leaf  = (const float*)d_in[3];
  const float* gmma  = (const float*)d_in[4];
  const float* beta  = (const float*)d_in[5];
  const float* mean  = (const float*)d_in[6];
  const float* var   = (const float*)d_in[7];
  const float* fc_w  = (const float*)d_in[8];
  const float* fc_b  = (const float*)d_in[9];
  float* out = (float*)d_out;

  // ---- workspace layout (no dec buffer anymore) ----
  char* ws = (char*)d_ws;
  __bf16* tnorm = (__bf16*)ws;                                   // 4.19 MB
  size_t  off   = (size_t)BATCH * NUM_TREES * sizeof(__bf16);
  __bf16* wdb   = (__bf16*)(ws + off);                           // 192 KB
  off += (size_t)NCOLS * INPUT_DIM * sizeof(__bf16);
  __bf16* fwb   = (__bf16*)(ws + off);                           // 32 KB

  // 0) one-shot weight conversions (L2-resident afterwards)
  cvt_bf16_kernel<<<(NCOLS * INPUT_DIM + 255) / 256, 256, 0, stream>>>(W_dec, wdb, NCOLS * INPUT_DIM);
  cvt_bf16_kernel<<<(INPUT_DIM * NUM_TREES + 255) / 256, 256, 0, stream>>>(fc_w, fwb, INPUT_DIM * NUM_TREES);

  // 1) fused decision GEMM + sigmoid + tree fold + BN
  const size_t lds_bytes = (size_t)64 * DSTRIDE * sizeof(float);   // 96.25 KB
  dec_tree_kernel<<<BATCH / 64, 256, lds_bytes, stream>>>(
      x, wdb, b_dec, leaf, gmma, beta, mean, var, tnorm);

  // 2) output GEMM + bias
  out_gemm_kernel<<<BATCH / 64, 256, 0, stream>>>(tnorm, fwb, fc_b, out);
}
